// Encoder_7748121002250
// MI455X (gfx1250) — matrix operand order repaired and checked by reading, not compiled, not run on hardware
//
#include <hip/hip_runtime.h>
#include <hip/hip_bf16.h>

#define N_NODES 2048
#define N_EDGES 4096
#define N_GRAPH 16
#define IN_DIM  11
#define HID     128
#define E_DIM   6
#define EDGE_H  64
#define EPS     1e-5f

// K windows per middle layer: 64*128/32 regular + 128/32 bias = 256 + 4 = 260
#define KWIN_REG 256
#define KWIN_ALL 260
#define BSW_PER_LAYER (KWIN_ALL * 8 * 32 * 16)   // elements (bf16)
#define KSPLIT 4                                  // waves per edge tile (k-range split)
#define KP (EDGE_H / KSPLIT)                      // k values per wave = 16

typedef __attribute__((ext_vector_type(16))) __bf16 v16bf;
typedef __attribute__((ext_vector_type(8)))  float  v8f;

__device__ __forceinline__ void atomAddF(float* p, float v) {
    unsafeAtomicAdd(p, v);   // -> global_atomic_add_f32
}

// ---------------------------------------------------------------- utilities
__global__ void k_zero(float* __restrict__ p, int n) {
    int i = blockIdx.x * blockDim.x + threadIdx.x;
    if (i < n) p[i] = 0.0f;
}

__global__ void k_deg(const int* __restrict__ dst, float* __restrict__ deg) {
    int e = blockIdx.x * blockDim.x + threadIdx.x;
    if (e < N_EDGES) atomAddF(&deg[dst[e]], 1.0f);
}

// h1[e,k] = relu(e_feat @ W1 + b1)   (E x 64)
__global__ void k_edge_mlp1(const float* __restrict__ ef,
                            const float* __restrict__ W1,
                            const float* __restrict__ b1,
                            float* __restrict__ h1) {
    int idx = blockIdx.x * blockDim.x + threadIdx.x;     // E*64 threads
    if (idx >= N_EDGES * EDGE_H) return;
    int e = idx >> 6, k = idx & 63;
    float acc = b1[k];
    #pragma unroll
    for (int d = 0; d < E_DIM; ++d)
        acc = fmaf(ef[e * E_DIM + d], W1[d * EDGE_H + k], acc);
    h1[idx] = fmaxf(acc, 0.0f);
}

// Convert W2/b2 (layers 1..3) to bf16, pre-swizzled into WMMA B-fragment order:
// Bsw[l][w][nt][lane][j];  lane<16: K=32w+j, col=lane;  lane>=16: K=32w+16+j, col=lane-16
__global__ void k_swizzleB(const float* __restrict__ eW2,
                           const float* __restrict__ eb2,
                           __hip_bfloat16* __restrict__ Bsw) {
    int idx = blockIdx.x * blockDim.x + threadIdx.x;
    if (idx >= 3 * BSW_PER_LAYER) return;
    int l   = idx / BSW_PER_LAYER;
    int rem = idx - l * BSW_PER_LAYER;
    int w    = rem >> 12;          // / 4096
    int nt   = (rem >> 9) & 7;
    int lane = (rem >> 4) & 31;
    int j    = rem & 15;
    int kk = w * 32 + j + ((lane >> 4) << 4);   // B-layout K offset
    int n  = nt * 16 + (lane & 15);
    float v;
    if (kk < EDGE_H * HID)
        v = eW2[(size_t)l * (EDGE_H * HID * HID) + (size_t)kk * HID + n];
    else
        v = eb2[(size_t)l * (HID * HID) + (size_t)(kk - EDGE_H * HID) * HID + n];
    Bsw[idx] = __float2bfloat16(v);
}

// Layer 0 (din = 11): VALU kernel, one edge per block, thread o = output channel
__global__ __launch_bounds__(HID)
void k_layer0_msg(const float* __restrict__ h, const float* __restrict__ h1,
                  const int* __restrict__ src, const int* __restrict__ dst,
                  const float* __restrict__ W2, const float* __restrict__ b2,
                  float* __restrict__ agg) {
    __shared__ float t[IN_DIM];
    __shared__ float u[EDGE_H];
    int e = blockIdx.x, o = threadIdx.x;
    if (o < IN_DIM)  t[o] = h[src[e] * IN_DIM + o];
    if (o < EDGE_H)  u[o] = h1[e * EDGE_H + o];
    __syncthreads();
    float acc = 0.0f;
    #pragma unroll 1
    for (int i = 0; i < IN_DIM; ++i) {
        float s = b2[i * HID + o];
        #pragma unroll 8
        for (int k = 0; k < EDGE_H; ++k)
            s = fmaf(u[k], W2[(k * IN_DIM + i) * HID + o], s);
        acc = fmaf(t[i], s, acc);
    }
    atomAddF(&agg[(size_t)dst[e] * HID + o], acc);
}

// Middle layers: fused bilinear message GEMM via WMMA bf16.
// Grid: (E/16) tiles x KSPLIT parts; one wave (32 threads) per (tile, part).
// Part p accumulates k in [p*16, p*16+16) plus bias window p, scatter-adds partials.
__global__ __launch_bounds__(32)
void k_msg_wmma(const float* __restrict__ x, const float* __restrict__ h1,
                const int* __restrict__ src, const int* __restrict__ dst,
                const __hip_bfloat16* __restrict__ Bsw, float* __restrict__ agg) {
    __shared__ float xs[16][HID];
    __shared__ float us[16][KP];
    __shared__ int   sdst[16];

    const int lane = threadIdx.x;
    const int tile = blockIdx.x >> 2;
    const int part = blockIdx.x & (KSPLIT - 1);
    const int e0   = tile * 16;
    const int kbeg = part * KP;

    for (int idx = lane; idx < 16 * HID; idx += 32) {
        int r = idx >> 7, c = idx & 127;
        xs[r][c] = x[(size_t)src[e0 + r] * HID + c];
    }
    for (int idx = lane; idx < 16 * KP; idx += 32) {
        int r = idx >> 4, c = idx & (KP - 1);
        us[r][c] = h1[(size_t)(e0 + r) * EDGE_H + kbeg + c];
    }
    if (lane < 16) sdst[lane] = dst[e0 + lane];
    __syncthreads();

    const int M  = lane & 15;   // edge row of A for this lane; output column n in C layout
    const int hv = lane >> 4;   // which K/M half

    v8f acc[8];
    #pragma unroll
    for (int nt = 0; nt < 8; ++nt)
        #pragma unroll
        for (int r = 0; r < 8; ++r) acc[nt][r] = 0.0f;

    const __hip_bfloat16* bbase = Bsw + (size_t)lane * 16;

    auto loadB = [&](v16bf* bf, int w) {
        const __hip_bfloat16* bw = bbase + (size_t)w * 4096;
        #pragma unroll
        for (int nt = 0; nt < 8; ++nt)
            bf[nt] = *(const v16bf*)(bw + nt * 512);
    };
    auto loadXF = [&](float* xf, int iw) {
        #pragma unroll
        for (int j = 0; j < 16; ++j) {
            int koff = j + hv * 16;   // A-layout K offset
            xf[j] = xs[M][iw * 32 + koff];
        }
    };
    auto doW = [&](const v16bf* bf, const float* xf, float u) {
        v16bf a;
        #pragma unroll
        for (int j = 0; j < 16; ++j) a[j] = (__bf16)(xf[j] * u);
        #pragma unroll
        for (int nt = 0; nt < 8; ++nt)
            acc[nt] = __builtin_amdgcn_wmma_f32_16x16x32_bf16(
                false, a, false, bf[nt], (short)0, acc[nt], false, false);
    };

    #pragma unroll 1
    for (int iw = 0; iw < 4; ++iw) {
        float xf[16];
        loadXF(xf, iw);
        v16bf b0[8], b1[8];
        loadB(b0, (kbeg + 0) * 4 + iw);          // prologue
        #pragma unroll 1
        for (int k = 0; k < KP; k += 2) {        // double-buffered pipeline
            loadB(b1, (kbeg + k + 1) * 4 + iw);
            doW(b0, xf, us[M][k]);
            if (k + 2 < KP) loadB(b0, (kbeg + k + 2) * 4 + iw);
            doW(b1, xf, us[M][k + 1]);
        }
    }
    // bias window for this part (u = 1): msg += x_s[:, part*32 .. +32) @ b2 rows
    {
        float xf[16];
        loadXF(xf, part);
        v16bf b0[8];
        loadB(b0, KWIN_REG + part);
        doW(b0, xf, 1.0f);
    }
    // Epilogue: C layout -> scatter-add partial sums into agg[dst]
    #pragma unroll
    for (int nt = 0; nt < 8; ++nt) {
        int n = nt * 16 + M;
        #pragma unroll
        for (int r = 0; r < 8; ++r) {
            int er = r + 8 * hv;
            atomAddF(&agg[(size_t)sdst[er] * HID + n], acc[nt][r]);
        }
    }
}

// mean-agg + NNConv bias -> relu -> BN(eval)
__global__ void k_node_post(const float* __restrict__ agg, const float* __restrict__ deg,
                            const float* __restrict__ bias, const float* __restrict__ ga,
                            const float* __restrict__ be, const float* __restrict__ mu,
                            const float* __restrict__ vv, float* __restrict__ xout) {
    int idx = blockIdx.x * blockDim.x + threadIdx.x;
    if (idx >= N_NODES * HID) return;
    int n = idx >> 7, o = idx & 127;
    float v = agg[idx] / fmaxf(deg[n], 1.0f) + bias[o];
    v = fmaxf(v, 0.0f);
    xout[idx] = (v - mu[o]) * rsqrtf(vv[o] + EPS) * ga[o] + be[o];
}

// per-graph mean (then BN) and max readout
__global__ __launch_bounds__(HID)
void k_readout(const float* __restrict__ x, const int* __restrict__ ng,
               const float* __restrict__ g_out, const float* __restrict__ b_out,
               const float* __restrict__ m_out, const float* __restrict__ v_out,
               float* __restrict__ out) {
    int g = blockIdx.x, o = threadIdx.x;
    float sum = 0.0f, mx = -3.4e38f, cnt = 0.0f;
    for (int n = 0; n < N_NODES; ++n) {
        if (ng[n] == g) {
            float v = x[(size_t)n * HID + o];
            sum += v; mx = fmaxf(mx, v); cnt += 1.0f;
        }
    }
    float hn = sum / cnt;
    hn = (hn - m_out[o]) * rsqrtf(v_out[o] + EPS) * g_out[o] + b_out[o];
    out[g * 2 * HID + o]       = hn;
    out[g * 2 * HID + HID + o] = mx;
}

extern "C" void kernel_launch(void* const* d_in, const int* in_sizes, int n_in,
                              void* d_out, int out_size, void* d_ws, size_t ws_size,
                              hipStream_t stream) {
    const float* h       = (const float*)d_in[0];
    const float* ef      = (const float*)d_in[1];
    const int*   src     = (const int*)d_in[2];
    const int*   dst     = (const int*)d_in[3];
    const int*   ng      = (const int*)d_in[4];
    const float* eW1_0   = (const float*)d_in[5];
    const float* eb1_0   = (const float*)d_in[6];
    const float* eW2_0   = (const float*)d_in[7];
    const float* eb2_0   = (const float*)d_in[8];
    const float* bias_0  = (const float*)d_in[9];
    const float* gamma_0 = (const float*)d_in[10];
    const float* beta_0  = (const float*)d_in[11];
    const float* mean_0  = (const float*)d_in[12];
    const float* var_0   = (const float*)d_in[13];
    const float* eW1     = (const float*)d_in[14];
    const float* eb1     = (const float*)d_in[15];
    const float* eW2     = (const float*)d_in[16];
    const float* eb2     = (const float*)d_in[17];
    const float* bias    = (const float*)d_in[18];
    const float* gamma   = (const float*)d_in[19];
    const float* beta    = (const float*)d_in[20];
    const float* mean    = (const float*)d_in[21];
    const float* var     = (const float*)d_in[22];
    const float* g_out   = (const float*)d_in[23];
    const float* b_out   = (const float*)d_in[24];
    const float* m_out   = (const float*)d_in[25];
    const float* v_out   = (const float*)d_in[26];

    // workspace layout
    float* x_cur = (float*)d_ws;                       // N*HID
    float* h1    = x_cur + N_NODES * HID;              // E*EDGE_H
    float* agg   = h1 + N_EDGES * EDGE_H;              // N*HID
    float* deg   = agg + N_NODES * HID;                // N
    __hip_bfloat16* Bsw = (__hip_bfloat16*)(deg + N_NODES);  // 3*BSW_PER_LAYER bf16

    const int NH = N_NODES * HID;

    // degrees (depend only on dst)
    k_zero<<<(N_NODES + 255) / 256, 256, 0, stream>>>(deg, N_NODES);
    k_deg<<<(N_EDGES + 255) / 256, 256, 0, stream>>>(dst, deg);
    // pre-swizzle W2/b2 for layers 1..3 into bf16 B-fragments
    k_swizzleB<<<(3 * BSW_PER_LAYER + 255) / 256, 256, 0, stream>>>(eW2, eb2, Bsw);

    // ---- layer 0 (din = 11, VALU path) ----
    k_edge_mlp1<<<(N_EDGES * EDGE_H + 255) / 256, 256, 0, stream>>>(ef, eW1_0, eb1_0, h1);
    k_zero<<<(NH + 255) / 256, 256, 0, stream>>>(agg, NH);
    k_layer0_msg<<<N_EDGES, HID, 0, stream>>>(h, h1, src, dst, eW2_0, eb2_0, agg);
    k_node_post<<<(NH + 255) / 256, 256, 0, stream>>>(agg, deg, bias_0, gamma_0,
                                                      beta_0, mean_0, var_0, x_cur);

    // ---- layers 1..3 (din = 128, WMMA path) ----
    for (int l = 0; l < 3; ++l) {
        k_edge_mlp1<<<(N_EDGES * EDGE_H + 255) / 256, 256, 0, stream>>>(
            ef, eW1 + (size_t)l * E_DIM * EDGE_H, eb1 + (size_t)l * EDGE_H, h1);
        k_zero<<<(NH + 255) / 256, 256, 0, stream>>>(agg, NH);
        k_msg_wmma<<<(N_EDGES / 16) * KSPLIT, 32, 0, stream>>>(
            x_cur, h1, src, dst, Bsw + (size_t)l * BSW_PER_LAYER, agg);
        k_node_post<<<(NH + 255) / 256, 256, 0, stream>>>(
            agg, deg, bias + l * HID, gamma + l * HID, beta + l * HID,
            mean + l * HID, var + l * HID, x_cur);
    }

    // ---- readout ----
    k_readout<<<N_GRAPH, HID, 0, stream>>>(x_cur, ng, g_out, b_out, m_out, v_out,
                                           (float*)d_out);
}
